// RoPEMultiHeadAttention_33681133535755
// MI455X (gfx1250) — matrix operand order repaired and checked by reading, not compiled, not run on hardware
//
#include <hip/hip_runtime.h>
#include <hip/hip_bf16.h>
#include <math.h>
#include <stdint.h>

typedef __attribute__((ext_vector_type(16))) _Float16 v16h;
typedef __attribute__((ext_vector_type(8)))  float    v8f;
typedef __attribute__((ext_vector_type(4)))  int      v4i;

#define BATCH    4
#define SEQ      2048
#define DMODEL   1024
#define NHEADS   16
#define HDIM     64

static __device__ inline v8f wmma_f16(v16h a, v16h b, v8f c) {
  // 8 args: (neg_a, A, neg_b, B, c_mod, C, reuse_a, reuse_b)
  return __builtin_amdgcn_wmma_f32_16x16x32_f16(false, a, false, b, (short)0, c,
                                                false, false);
}

// --------------------------------------------------------------------------
// CDNA5 async global->LDS copy path (guarded: falls back to load+ds_store)
// --------------------------------------------------------------------------
#if defined(__gfx1250__) &&                                                   \
    __has_builtin(__builtin_amdgcn_global_load_async_to_lds_b128) &&          \
    __has_builtin(__builtin_amdgcn_s_wait_asynccnt)
#define USE_ASYNC_LDS 1
#else
#define USE_ASYNC_LDS 0
#endif

#if USE_ASYNC_LDS
typedef __attribute__((address_space(1))) v4i as1_v4i;
typedef __attribute__((address_space(3))) v4i as3_v4i;
#endif

// Copy 16 bytes global -> LDS (per calling lane).
static __device__ inline void copy_b128_to_lds(const void* gsrc, void* ldst) {
#if USE_ASYNC_LDS
  // Flat->AS1 pointers are value-identical; low 32 bits of a flat LDS
  // pointer are the LDS offset (AS3 is 32-bit).
  __builtin_amdgcn_global_load_async_to_lds_b128(
      (as1_v4i*)(uintptr_t)gsrc, (as3_v4i*)(uint32_t)(uintptr_t)ldst, 0, 0);
#else
  *(uint4*)ldst = *(const uint4*)gsrc;
#endif
}

static __device__ inline void lds_copy_fence() {
#if USE_ASYNC_LDS
  __builtin_amdgcn_s_wait_asynccnt(0);
#endif
}

// ---------------------------------------------------------------------------
// f32 -> f16 conversion (grid-stride)
// ---------------------------------------------------------------------------
__global__ void cvt_f32_to_f16_kernel(const float* __restrict__ src,
                                      _Float16* __restrict__ dst, int n) {
  int i = blockIdx.x * blockDim.x + threadIdx.x;
  int stride = gridDim.x * blockDim.x;
  for (; i < n; i += stride) dst[i] = (_Float16)src[i];
}

// ---------------------------------------------------------------------------
// Tiled WMMA GEMM: C[M,N] = A[M,K](f16) * B[K,N](f16) + bias
// Block: 256 threads (8 waves), tile 128(M) x 64(N), K-step 32.
// Wave (wm 0..3, wn 0..1) computes 32x32 via 2x2 WMMA 16x16x32 tiles.
// ---------------------------------------------------------------------------
template <bool OUT_F16>
__global__ void __launch_bounds__(256)
gemm_f16_kernel(const _Float16* __restrict__ A, const _Float16* __restrict__ B,
                const float* __restrict__ bias, void* __restrict__ Cout,
                int M, int N, int K) {
  constexpr int AS = 40;  // padded LDS stride for A tile rows (f16)
  constexpr int BS = 36;  // padded LDS stride for BT rows (f16)
  __shared__ _Float16 As[128 * AS];
  __shared__ _Float16 BTs[64 * BS];

  const int tid  = threadIdx.x;
  const int lane = tid & 31;
  const int wave = tid >> 5;
  const int wm   = wave >> 1;   // 0..3
  const int wn   = wave & 1;    // 0..1
  const int half = lane >> 4;
  const int ln   = lane & 15;
  const int m0   = blockIdx.y * 128;
  const int n0   = blockIdx.x * 64;

  v8f zero = {};
  v8f acc[2][2];
#pragma unroll
  for (int i = 0; i < 2; ++i)
#pragma unroll
    for (int j = 0; j < 2; ++j) acc[i][j] = zero;

  for (int kk = 0; kk < K; kk += 32) {
    __syncthreads();
    // A tile: 128 rows x 32 cols = 512 b128 chunks, async -> LDS
#pragma unroll 2
    for (int s = tid; s < 512; s += 256) {
      int r = s >> 2, c = s & 3;
      copy_b128_to_lds(A + (size_t)(m0 + r) * K + kk + c * 8,
                       &As[r * AS + c * 8]);
    }
    // B tile transposed into BTs[n][k]: 32 k-rows x 64 n-cols (needs VGPR path)
    {
      int s = tid;  // 256 slots, one per thread
      int kr = s >> 3, n8 = s & 7;
      uint4 v = *(const uint4*)(B + (size_t)(kk + kr) * N + n0 + n8 * 8);
      const _Float16* pf = (const _Float16*)&v;
#pragma unroll
      for (int j = 0; j < 8; ++j) BTs[(n8 * 8 + j) * BS + kr] = pf[j];
    }
    lds_copy_fence();
    __syncthreads();

    // B fragments (shared across mt)
    v16h bf[2];
#pragma unroll
    for (int nt = 0; nt < 2; ++nt) {
      const _Float16* brow = &BTs[(wn * 32 + nt * 16 + ln) * BS];
#pragma unroll
      for (int i = 0; i < 16; ++i) bf[nt][i] = brow[(i < 8 ? 0 : 8) + half * 8 + i];
    }
#pragma unroll
    for (int mt = 0; mt < 2; ++mt) {
      v16h a;
      const _Float16* arow = &As[(wm * 32 + mt * 16 + ln) * AS];
#pragma unroll
      for (int i = 0; i < 8; ++i) a[i] = arow[half * 8 + i];          // K 0..15
#pragma unroll
      for (int i = 0; i < 8; ++i) a[8 + i] = arow[16 + half * 8 + i]; // K 16..31
#pragma unroll
      for (int nt = 0; nt < 2; ++nt) acc[mt][nt] = wmma_f16(a, bf[nt], acc[mt][nt]);
    }
  }

  // Epilogue: C/D layout -> row = g + half*8, col = ln (per 16x16 tile)
#pragma unroll
  for (int mt = 0; mt < 2; ++mt) {
#pragma unroll
    for (int nt = 0; nt < 2; ++nt) {
      int col = n0 + wn * 32 + nt * 16 + ln;
      float bv = bias[col];
#pragma unroll
      for (int g = 0; g < 8; ++g) {
        int row = m0 + wm * 32 + mt * 16 + half * 8 + g;
        float v = acc[mt][nt][g] + bv;
        if (OUT_F16)
          ((_Float16*)Cout)[(size_t)row * N + col] = (_Float16)v;
        else
          ((float*)Cout)[(size_t)row * N + col] = v;
      }
    }
  }
}

// ---------------------------------------------------------------------------
// RoPE applied in place to Q and K slices of qkv (f16), layout [B*S][3*1024]
// with column = mat*1024 + h*64 + d.
// ---------------------------------------------------------------------------
__global__ void rope_kernel(_Float16* __restrict__ qkv) {
  int idx = blockIdx.x * blockDim.x + threadIdx.x;  // B*S*H*32 work items
  int i = idx & 31;              // rotation pair index
  int h = (idx >> 5) & (NHEADS - 1);
  int row = idx >> 9;            // b*S + s
  if (row >= BATCH * SEQ) return;
  int s = row & (SEQ - 1);
  // inv_freq = 10000^(-i/32) = exp(-i * ln(10000)/32)
  float inv = __expf(-(float)i * 0.28782313662425572f);
  float ang = (float)s * inv;
  float c = __cosf(ang), sn = __sinf(ang);
#pragma unroll
  for (int m = 0; m < 2; ++m) {  // m=0: Q, m=1: K
    _Float16* base = qkv + (size_t)row * (3 * DMODEL) + m * DMODEL + h * HDIM;
    float x1 = (float)base[i];
    float x2 = (float)base[i + 32];
    base[i]      = (_Float16)(x1 * c - x2 * sn);
    base[i + 32] = (_Float16)(x2 * c + x1 * sn);
  }
}

// ---------------------------------------------------------------------------
// Flash attention: block = 128 threads (4 waves), 64 query rows per block,
// one (b,h) per blockIdx.y. Key blocks of 64 (16 WMMA per block iteration),
// online softmax in f32 with wave32 half-group reductions.
// ---------------------------------------------------------------------------
__global__ void __launch_bounds__(128)
attn_kernel(const _Float16* __restrict__ qkv, _Float16* __restrict__ outh) {
  constexpr int KB = 64;  // keys per block iteration
  constexpr int KS = 72;  // Ks row stride (64 head dims + pad, rows 16B-aligned)
  constexpr int VS = 72;  // VTs row stride (64 keys + pad)
  constexpr int PS = 72;  // Ps row stride (64 keys + pad)
  __shared__ _Float16 Ks[KB * KS];     // [key][head]
  __shared__ _Float16 VTs[64 * VS];    // [head][key]
  __shared__ _Float16 Ps[4][16 * PS];  // per wave: [q-row][key]

  const int tid  = threadIdx.x;
  const int lane = tid & 31;
  const int wave = tid >> 5;
  const int half = lane >> 4;
  const int ln   = lane & 15;
  const int bh = blockIdx.y;
  const int b  = bh >> 4;
  const int h  = bh & (NHEADS - 1);
  const int q0 = blockIdx.x * 64;

  // Q A-fragments for this wave's 16 rows (head-dim chunks of 32)
  v16h qa[2];
  {
    const _Float16* qrow =
        qkv + (size_t)(b * SEQ + q0 + wave * 16 + ln) * (3 * DMODEL) + h * HDIM;
#pragma unroll
    for (int c = 0; c < 2; ++c) {
#pragma unroll
      for (int i = 0; i < 8; ++i) qa[c][i] = qrow[c * 32 + half * 8 + i];
#pragma unroll
      for (int i = 0; i < 8; ++i) qa[c][8 + i] = qrow[c * 32 + 16 + half * 8 + i];
    }
  }

  v8f zero = {};
  v8f o[4];
#pragma unroll
  for (int nt = 0; nt < 4; ++nt) o[nt] = zero;
  float row_max[8], row_sum[8];
#pragma unroll
  for (int g = 0; g < 8; ++g) { row_max[g] = -1e30f; row_sum[g] = 0.0f; }

  const float scale = 0.125f;  // 1/sqrt(64)

  for (int kb = 0; kb < SEQ / KB; ++kb) {
    const size_t krow0 = (size_t)(b * SEQ + kb * KB) * (3 * DMODEL);
    __syncthreads();
    // K tile: 64 keys x 64 dims, row-major [key][head], async -> LDS
#pragma unroll 4
    for (int s = tid; s < 512; s += 128) {
      int key = s >> 3, c8 = s & 7;
      copy_b128_to_lds(qkv + krow0 + (size_t)key * (3 * DMODEL) + DMODEL +
                           h * HDIM + c8 * 8,
                       &Ks[key * KS + c8 * 8]);
    }
    // V tile transposed into VTs[head][key] (VGPR path: transpose on store)
#pragma unroll 4
    for (int s = tid; s < 512; s += 128) {
      int key = s >> 3, c8 = s & 7;
      uint4 v = *(const uint4*)(qkv + krow0 + (size_t)key * (3 * DMODEL) +
                                2 * DMODEL + h * HDIM + c8 * 8);
      const _Float16* pf = (const _Float16*)&v;
#pragma unroll
      for (int j = 0; j < 8; ++j) VTs[(c8 * 8 + j) * VS + key] = pf[j];
    }
    // L2 prefetch of the next key block's K/V stream
    if (kb + 1 < SEQ / KB) {
      const char* nxt = (const char*)(qkv + krow0 +
                                      (size_t)KB * (3 * DMODEL) + DMODEL);
      __builtin_prefetch(nxt + tid * 128, 0, 3);
    }
    lds_copy_fence();
    __syncthreads();

    // Scores: 16 q-rows x 64 keys = four 16x16 WMMA tiles, K-dim = head 64
    v8f sc[4];
#pragma unroll
    for (int nt = 0; nt < 4; ++nt) {
      v8f c = zero;
#pragma unroll
      for (int ch = 0; ch < 2; ++ch) {
        v16h bfr;  // B frag: k = head (ch*32 + (i<8?0:8) + half*8 + i), n = key (ln)
        const _Float16* kr = &Ks[(nt * 16 + ln) * KS + ch * 32];
#pragma unroll
        for (int i = 0; i < 16; ++i) bfr[i] = kr[(i < 8 ? 0 : 8) + half * 8 + i];
        c = wmma_f16(qa[ch], bfr, c);
      }
      sc[nt] = c;
    }

    // Online softmax; row g lives in lanes {half*16 .. half*16+15}
#pragma unroll
    for (int g = 0; g < 8; ++g) {
      float v0 = sc[0][g] * scale;
      float v1 = sc[1][g] * scale;
      float v2 = sc[2][g] * scale;
      float v3 = sc[3][g] * scale;
      float mloc = fmaxf(fmaxf(v0, v1), fmaxf(v2, v3));
#pragma unroll
      for (int off = 1; off < 16; off <<= 1)
        mloc = fmaxf(mloc, __shfl_xor(mloc, off, 32));
      float mnew = fmaxf(row_max[g], mloc);
      float corr = __expf(row_max[g] - mnew);
      row_max[g] = mnew;
      float p0 = __expf(v0 - mnew);
      float p1 = __expf(v1 - mnew);
      float p2 = __expf(v2 - mnew);
      float p3 = __expf(v3 - mnew);
      float sloc = (p0 + p1) + (p2 + p3);
#pragma unroll
      for (int off = 1; off < 16; off <<= 1)
        sloc += __shfl_xor(sloc, off, 32);
      row_sum[g] = row_sum[g] * corr + sloc;
#pragma unroll
      for (int nt = 0; nt < 4; ++nt) o[nt][g] = o[nt][g] * corr;
      // Store P tile [q-row][key] (f16) for the P*V A-fragments
      _Float16* prw = &Ps[wave][(half * 8 + g) * PS];
      prw[ln]      = (_Float16)p0;
      prw[16 + ln] = (_Float16)p1;
      prw[32 + ln] = (_Float16)p2;
      prw[48 + ln] = (_Float16)p3;
    }
    __syncthreads();

    // P (16x64) * V (64x64): 4 head tiles x 2 key chunks = 8 WMMA
    v16h pa[2];
    const _Float16* prow = &Ps[wave][ln * PS];
#pragma unroll
    for (int ch = 0; ch < 2; ++ch) {
#pragma unroll
      for (int i = 0; i < 8; ++i) pa[ch][i] = prow[ch * 32 + half * 8 + i];
#pragma unroll
      for (int i = 0; i < 8; ++i) pa[ch][8 + i] = prow[ch * 32 + 16 + half * 8 + i];
    }
#pragma unroll
    for (int nt = 0; nt < 4; ++nt) {
#pragma unroll
      for (int ch = 0; ch < 2; ++ch) {
        v16h bfr;  // B frag: k = key (ch*32 + (i<8?0:8) + half*8 + i), n = head (nt*16+ln)
        const _Float16* vr = &VTs[(nt * 16 + ln) * VS + ch * 32];
#pragma unroll
        for (int i = 0; i < 16; ++i) bfr[i] = vr[(i < 8 ? 0 : 8) + half * 8 + i];
        o[nt] = wmma_f16(pa[ch], bfr, o[nt]);
      }
    }
  }

  // Normalize and write out in (b, s, h, d) -> [B*S][1024] layout
#pragma unroll
  for (int nt = 0; nt < 4; ++nt) {
#pragma unroll
    for (int g = 0; g < 8; ++g) {
      int row = q0 + wave * 16 + half * 8 + g;
      float v = o[nt][g] / row_sum[g];
      outh[(size_t)(b * SEQ + row) * DMODEL + h * HDIM + nt * 16 + ln] = (_Float16)v;
    }
  }
}

// ---------------------------------------------------------------------------
// Launcher
// ---------------------------------------------------------------------------
extern "C" void kernel_launch(void* const* d_in, const int* in_sizes, int n_in,
                              void* d_out, int out_size, void* d_ws, size_t ws_size,
                              hipStream_t stream) {
  const float* x    = (const float*)d_in[0];
  const float* Wqkv = (const float*)d_in[1];
  const float* bqkv = (const float*)d_in[2];
  const float* Wout = (const float*)d_in[3];
  const float* bout = (const float*)d_in[4];
  float* out = (float*)d_out;

  const int M = BATCH * SEQ;  // 8192

  char* ws = (char*)d_ws;
  size_t off = 0;
  auto wsalloc = [&](size_t bytes) {
    void* p = ws + off;
    off = (off + bytes + 255) & ~(size_t)255;
    return p;
  };
  _Float16* xh    = (_Float16*)wsalloc((size_t)M * DMODEL * 2);          // 16 MB
  _Float16* wqkvh = (_Float16*)wsalloc((size_t)DMODEL * 3 * DMODEL * 2); // 6 MB
  _Float16* wouth = (_Float16*)wsalloc((size_t)DMODEL * DMODEL * 2);     // 2 MB
  _Float16* qkvh  = (_Float16*)wsalloc((size_t)M * 3 * DMODEL * 2);      // 48 MB
  _Float16* attnh = (_Float16*)wsalloc((size_t)M * DMODEL * 2);          // 16 MB
  (void)ws_size; (void)in_sizes; (void)n_in; (void)out_size;

  // 1) f32 -> f16 conversions
  cvt_f32_to_f16_kernel<<<2048, 256, 0, stream>>>(x, xh, M * DMODEL);
  cvt_f32_to_f16_kernel<<<1024, 256, 0, stream>>>(Wqkv, wqkvh, DMODEL * 3 * DMODEL);
  cvt_f32_to_f16_kernel<<<512, 256, 0, stream>>>(Wout, wouth, DMODEL * DMODEL);

  // 2) QKV projection: [8192,1024] x [1024,3072] + bqkv -> f16
  gemm_f16_kernel<true><<<dim3(3 * DMODEL / 64, M / 128), 256, 0, stream>>>(
      xh, wqkvh, bqkv, qkvh, M, 3 * DMODEL, DMODEL);

  // 3) RoPE on Q and K
  rope_kernel<<<(M * NHEADS * 32) / 256, 256, 0, stream>>>(qkvh);

  // 4) Flash attention per (b,h), 64 query rows per block
  attn_kernel<<<dim3(SEQ / 64, BATCH * NHEADS), 128, 0, stream>>>(qkvh, attnh);

  // 5) Output projection: [8192,1024] x [1024,1024] + bout -> f32
  gemm_f16_kernel<false><<<dim3(DMODEL / 64, M / 128), 256, 0, stream>>>(
      attnh, wouth, bout, out, M, DMODEL, DMODEL);
}
